// LSQLinear_7404523618310
// MI455X (gfx1250) — matrix operand order repaired and checked by reading, not compiled, not run on hardware
//
#include <hip/hip_runtime.h>

typedef unsigned short u16;
typedef u16   v8u  __attribute__((ext_vector_type(8)));
typedef __bf16 bf16_t;
typedef bf16_t v16bf __attribute__((ext_vector_type(16)));
typedef float v8f __attribute__((ext_vector_type(8)));
typedef float v4f __attribute__((ext_vector_type(4)));
typedef int   v4i __attribute__((ext_vector_type(4)));

#define M_TOT 8192
#define N_TOT 4096
#define K_TOT 4096
#define NGRP  32      // I / 128
#define BLK   128
#define BK    64
#define LDSK  72      // padded row stride (halfwords); 144B, 16B-aligned
#define TILE_HW (BLK * LDSK)

#if __has_builtin(__builtin_amdgcn_global_load_async_to_lds_b128)
#define USE_ASYNC 1
#else
#define USE_ASYNC 0
#endif

#define ASG __attribute__((address_space(1)))
#define ASL __attribute__((address_space(3)))

__device__ __forceinline__ u16 f2bf(float f) {
  unsigned u = __float_as_uint(f);
  u += 0x7FFFu + ((u >> 16) & 1u);   // round-to-nearest-even
  return (u16)(u >> 16);
}

// ---------------- x: f32 -> bf16 ----------------
__global__ __launch_bounds__(256) void cvt_x(const float* __restrict__ x,
                                             u16* __restrict__ xb) {
  size_t base = ((size_t)blockIdx.x * 256 + threadIdx.x) * 8;
  v4f a = *(const v4f*)(x + base);
  v4f b = *(const v4f*)(x + base + 4);
  v8u o;
  o[0] = f2bf(a[0]); o[1] = f2bf(a[1]); o[2] = f2bf(a[2]); o[3] = f2bf(a[3]);
  o[4] = f2bf(b[0]); o[5] = f2bf(b[1]); o[6] = f2bf(b[2]); o[7] = f2bf(b[3]);
  *(v8u*)(xb + base) = o;
}

// ------------- weights: int8 codes -> bf16 dequant -------------
__global__ __launch_bounds__(256) void dequant_w(const int* __restrict__ q,
                                                 const float* __restrict__ scale,
                                                 const float* __restrict__ off,
                                                 u16* __restrict__ wb) {
  size_t base = ((size_t)blockIdx.x * 256 + threadIdx.x) * 8;
  int o = (int)(base >> 12);       // / 4096
  int i = (int)(base & 4095);
  int g = i >> 7;                  // group of 128 (8 consecutive never cross)
  float s = scale[o * NGRP + g];
  float f0 = off[o * NGRP + g];
  v4i q0 = *(const v4i*)(q + base);
  v4i q1 = *(const v4i*)(q + base + 4);
  v8u r;
  r[0] = f2bf((float)q0[0] * s - f0); r[1] = f2bf((float)q0[1] * s - f0);
  r[2] = f2bf((float)q0[2] * s - f0); r[3] = f2bf((float)q0[3] * s - f0);
  r[4] = f2bf((float)q1[0] * s - f0); r[5] = f2bf((float)q1[1] * s - f0);
  r[6] = f2bf((float)q1[2] * s - f0); r[7] = f2bf((float)q1[3] * s - f0);
  *(v8u*)(wb + base) = r;
}

#if USE_ASYNC
__device__ __forceinline__ void async_b128(const u16* g, const u16* l) {
  __builtin_amdgcn_global_load_async_to_lds_b128(
      (ASG v4i*)(size_t)g, (ASL v4i*)(unsigned)(size_t)l, 0, 0);
}
#endif

// ------------- GEMM: C[M,N] = A[M,K] * B[N,K]^T, bf16 WMMA, f32 acc -------------
__global__ __launch_bounds__(256) void gemm_bf16(const u16* __restrict__ A,
                                                 const u16* __restrict__ B,
                                                 float* __restrict__ C) {
  __shared__ u16 sA[2][TILE_HW];
  __shared__ u16 sB[2][TILE_HW];

  const int t    = threadIdx.x;
  const int lane = t & 31;
  const int wid  = t >> 5;
  const int bm   = blockIdx.y;
  const int bn   = blockIdx.x;
  const int wm   = (wid & 3) * 32;   // wave row offset inside 128-tile
  const int wn   = (wid >> 2) * 64;  // wave col offset inside 128-tile
  const int lh   = lane >> 4;        // lane half (0/1)
  const int lm   = lane & 15;

  v8f acc[2][4] = {};

  const u16* aBase = A + (size_t)(bm * BLK) * K_TOT;
  const u16* bBase = B + (size_t)(bn * BLK) * K_TOT;
  const int NK = K_TOT / BK;

#if USE_ASYNC
  // preload tile 0 via async DMA to LDS
#pragma unroll
  for (int i = 0; i < 4; ++i) {
    int chunk = t + 256 * i;
    int row   = chunk >> 3;
    int cc    = (chunk & 7) * 8;
    async_b128(aBase + (size_t)row * K_TOT + cc, &sA[0][row * LDSK + cc]);
    async_b128(bBase + (size_t)row * K_TOT + cc, &sB[0][row * LDSK + cc]);
  }
#else
  v8u ra[4], rb[4];
#pragma unroll
  for (int i = 0; i < 4; ++i) {
    int chunk = t + 256 * i;
    int row   = chunk >> 3;
    int cc    = (chunk & 7) * 8;
    ra[i] = *(const v8u*)(aBase + (size_t)row * K_TOT + cc);
    rb[i] = *(const v8u*)(bBase + (size_t)row * K_TOT + cc);
  }
#pragma unroll
  for (int i = 0; i < 4; ++i) {
    int chunk = t + 256 * i;
    int row   = chunk >> 3;
    int cc    = (chunk & 7) * 8;
    *(v8u*)&sA[0][row * LDSK + cc] = ra[i];
    *(v8u*)&sB[0][row * LDSK + cc] = rb[i];
  }
#endif

  for (int it = 0; it < NK; ++it) {
    const int cur = it & 1;
    const int nxt = cur ^ 1;
    const int ktn = (it + 1) * BK;

#if USE_ASYNC
    // wait for this tile's DMA, make it visible workgroup-wide, then kick
    // off the next tile's DMA into the other buffer (overlaps with WMMAs).
    asm volatile("s_wait_asynccnt 0" ::: "memory");
    __syncthreads();
    if (it + 1 < NK) {
#pragma unroll
      for (int i = 0; i < 4; ++i) {
        int chunk = t + 256 * i;
        int row   = chunk >> 3;
        int cc    = (chunk & 7) * 8;
        async_b128(aBase + (size_t)row * K_TOT + ktn + cc,
                   &sA[nxt][row * LDSK + cc]);
        async_b128(bBase + (size_t)row * K_TOT + ktn + cc,
                   &sB[nxt][row * LDSK + cc]);
      }
    }
#else
    __syncthreads();
    if (it + 1 < NK) {
#pragma unroll
      for (int i = 0; i < 4; ++i) {
        int chunk = t + 256 * i;
        int row   = chunk >> 3;
        int cc    = (chunk & 7) * 8;
        ra[i] = *(const v8u*)(aBase + (size_t)row * K_TOT + ktn + cc);
        rb[i] = *(const v8u*)(bBase + (size_t)row * K_TOT + ktn + cc);
        if (i == 0 && it + 2 < NK) {
          __builtin_prefetch(aBase + (size_t)row * K_TOT + ktn + BK + cc, 0, 0);
          __builtin_prefetch(bBase + (size_t)row * K_TOT + ktn + BK + cc, 0, 0);
        }
      }
    }
#endif

    // ---- compute on buf[cur] ----
#pragma unroll
    for (int ks = 0; ks < BK; ks += 32) {
      union Frag { v8u h[2]; v16bf v; } af[2], bfr[4];
#pragma unroll
      for (int mt = 0; mt < 2; ++mt) {
        const u16* p = &sA[cur][(wm + mt * 16 + lm) * LDSK + ks + lh * 8];
        af[mt].h[0] = *(const v8u*)p;
        af[mt].h[1] = *(const v8u*)(p + 16);
      }
#pragma unroll
      for (int nt = 0; nt < 4; ++nt) {
        const u16* p = &sB[cur][(wn + nt * 16 + lm) * LDSK + ks + lh * 8];
        bfr[nt].h[0] = *(const v8u*)p;
        bfr[nt].h[1] = *(const v8u*)(p + 16);
      }
#pragma unroll
      for (int mt = 0; mt < 2; ++mt)
#pragma unroll
        for (int nt = 0; nt < 4; ++nt)
          acc[mt][nt] = __builtin_amdgcn_wmma_f32_16x16x32_bf16(
              false, af[mt].v, false, bfr[nt].v, (short)0, acc[mt][nt],
              false, false);
    }

#if !USE_ASYNC
    if (it + 1 < NK) {
#pragma unroll
      for (int i = 0; i < 4; ++i) {
        int chunk = t + 256 * i;
        int row   = chunk >> 3;
        int cc    = (chunk & 7) * 8;
        *(v8u*)&sA[nxt][row * LDSK + cc] = ra[i];
        *(v8u*)&sB[nxt][row * LDSK + cc] = rb[i];
      }
    }
#endif
  }

  // writeout: VGPR r holds (m = r + 8*lh, n = lm) of each 16x16 tile
#pragma unroll
  for (int mt = 0; mt < 2; ++mt) {
#pragma unroll
    for (int nt = 0; nt < 4; ++nt) {
      int col = bn * BLK + wn + nt * 16 + lm;
#pragma unroll
      for (int r = 0; r < 8; ++r) {
        int row = bm * BLK + wm + mt * 16 + lh * 8 + r;
        C[(size_t)row * N_TOT + col] = acc[mt][nt][r];
      }
    }
  }
}

extern "C" void kernel_launch(void* const* d_in, const int* in_sizes, int n_in,
                              void* d_out, int out_size, void* d_ws, size_t ws_size,
                              hipStream_t stream) {
  const float* x  = (const float*)d_in[0];
  const int*   qw = (const int*)d_in[1];
  const float* sc = (const float*)d_in[2];
  const float* of = (const float*)d_in[3];
  float* out = (float*)d_out;

  u16* xb = (u16*)d_ws;                         // 8192*4096 bf16 = 64 MiB
  u16* wb = xb + (size_t)M_TOT * K_TOT;         // 4096*4096 bf16 = 32 MiB

  cvt_x<<<(int)(((size_t)M_TOT * K_TOT / 8) / 256), 256, 0, stream>>>(x, xb);
  dequant_w<<<(int)(((size_t)N_TOT * K_TOT / 8) / 256), 256, 0, stream>>>(qw, sc, of, wb);
  gemm_bf16<<<dim3(N_TOT / BLK, M_TOT / BLK), 256, 0, stream>>>(xb, wb, out);
}
